// HypergraphNDP_4088808866137
// MI455X (gfx1250) — matrix operand order repaired and checked by reading, not compiled, not run on hardware
//
#include <hip/hip_runtime.h>
#include <hip/hip_bf16.h>

// Problem constants (from reference)
#define NTOT 8192
#define EDG  4096
#define DIM  256
#define HID  512
#define NH   (NTOT / 2)        // active nodes (node_mask = arange < N/2)
#define EMSK ((3 * EDG) / 4)   // active edges (edge_mask = arange < 3E/4)

typedef __bf16 bf16_t;
typedef __attribute__((ext_vector_type(16))) __bf16 v16bf;
typedef __attribute__((ext_vector_type(8)))  float  v8f;

__device__ __forceinline__ bf16_t f2bf(float f) { return (bf16_t)f; }

// ---------------------------------------------------------------------------
// WMMA bf16 GEMM:  C[M,Nc] = epilogue( A[M,K] @ B[K,Nc] ),  B given COLUMN-
// MAJOR as Bc[Nc][K] (for y = x @ W.T this is just W[out][in] itself).
// Column-major B makes every lane's 32x16 B-fragment two contiguous 16-byte runs ->
// two global_load_b128, no LDS staging, no barriers in the K loop.
// Each wave computes a 16x(16*NT) strip: one A fragment feeds NT WMMAs.
// All NT B-fragments are loaded into distinct registers BEFORE the WMMA
// burst so the loads issue as one clause (max memory-level parallelism).
// Block = 128 threads = 4 waves covering 64 rows. Grid: (M/64, Nc/(16*NT)).
// M % 64 == 0, K % 32 == 0, Nc % (16*NT) == 0 for all calls here.
// ---------------------------------------------------------------------------
#define NT 4   // N-tiles per wave (16*NT = 64 output columns per wave)

__global__ void __launch_bounds__(128)
wmma_gemm_bf16(const bf16_t* __restrict__ A,
               const bf16_t* __restrict__ Bc,   // [Nc][K] col-major B
               int M, int K, int Nc,
               const float* __restrict__ row_scale,
               const float* __restrict__ bias,
               int do_relu,
               float* __restrict__ Cf, int ldcf,
               bf16_t* __restrict__ Cb, int ldcb, int colOffB, int storeT)
{
    const int lane = threadIdx.x & 31;
    const int wave = threadIdx.x >> 5;
    const int m0   = blockIdx.x * 64 + wave * 16;
    const int nb   = blockIdx.y * (16 * NT);

    // A-fragment (16x32 bf16, ISA layout):
    //   lanes 0-15 : M=lane,    elems 0..7 -> K 0..7,  8..15 -> K 16..23
    //   lanes16-31 : M=lane-16, elems 0..7 -> K 8..15, 8..15 -> K 24..31
    const int aRow = m0 + (lane & 15);
    const int aKlo = (lane < 16) ? 0 : 8;
    // B-fragment (32x16): lanes 0-15: N=lane, K 0..7,16..23; lanes16-31: N=lane-16, K 8..15,24..31
    const int colIT = lane & 15;
    const int bKlo  = (lane < 16) ? 0 : 8;

    v8f acc[NT];
#pragma unroll
    for (int t = 0; t < NT; ++t) acc[t] = (v8f){};

    const bf16_t* aBase = A + (size_t)aRow * K + aKlo;
    const bf16_t* bBase[NT];
#pragma unroll
    for (int t = 0; t < NT; ++t)
        bBase[t] = Bc + (size_t)(nb + t * 16 + colIT) * K + bKlo;

    for (int k0 = 0; k0 < K; k0 += 32) {
        union { v16bf v; uint4 q[2]; } afrag;
        union { v16bf v; uint4 q[2]; } bfrag[NT];

        // Issue all loads up front (one clause), then the WMMA burst.
        afrag.q[0] = *(const uint4*)(aBase + k0);
        afrag.q[1] = *(const uint4*)(aBase + k0 + 16);
#pragma unroll
        for (int t = 0; t < NT; ++t) {
            bfrag[t].q[0] = *(const uint4*)(bBase[t] + k0);
            bfrag[t].q[1] = *(const uint4*)(bBase[t] + k0 + 16);
        }
        if (k0 + 32 < K) __builtin_prefetch(aBase + k0 + 32, 0, 1);

#pragma unroll
        for (int t = 0; t < NT; ++t)
            acc[t] = __builtin_amdgcn_wmma_f32_16x16x32_bf16(
                false, afrag.v, false, bfrag[t].v, (short)0, acc[t], false, false);
    }

    // Epilogue. C/D layout: elem i -> row m0 + (lane<16 ? i : 8+i), col nb+t*16+(lane&15)
#pragma unroll
    for (int t = 0; t < NT; ++t) {
        const int col = nb + t * 16 + colIT;
        const float bv = bias ? bias[col] : 0.0f;
#pragma unroll
        for (int i = 0; i < 8; ++i) {
            int row = m0 + ((lane < 16) ? i : (8 + i));
            float v = acc[t][i];
            if (row_scale) v *= row_scale[row];
            v += bv;
            if (do_relu) v = fmaxf(v, 0.0f);
            if (Cf) Cf[(size_t)row * ldcf + col] = v;
            if (Cb) {
                if (storeT)  // store transposed: Cb[col][row]
                    Cb[(size_t)(colOffB + col) * ldcb + row] = f2bf(v);
                else
                    Cb[(size_t)row * ldcb + colOffB + col] = f2bf(v);
            }
        }
    }
}

// ---------------------------------------------------------------------------
// Tiled (32x32) transpose-convert f32 -> bf16 with optional masking and an
// optional row-major masked copy. Coalesced on both sides via LDS tile.
//   dstT[c*R + r] = mask(src[r*C + c]);  dstRM[r*ldRM + c] = same (if non-null)
// Grid: (C/32, R/32), block 256.
// ---------------------------------------------------------------------------
__global__ void __launch_bounds__(256)
tconv(const float* __restrict__ src, int R, int C,
      bf16_t* __restrict__ dstT,
      bf16_t* __restrict__ dstRM, int ldRM,
      int rowLim, int colLim)
{
    __shared__ bf16_t tile[32][33];
    const int c0 = blockIdx.x * 32, r0 = blockIdx.y * 32;
    const int tx = threadIdx.x & 31, ty = threadIdx.x >> 5;   // 32 x 8

#pragma unroll
    for (int i = 0; i < 4; ++i) {
        int r = r0 + ty + 8 * i, c = c0 + tx;
        float v = src[(size_t)r * C + c];
        if (r >= rowLim || c >= colLim) v = 0.0f;
        bf16_t b = f2bf(v);
        if (dstRM) dstRM[(size_t)r * ldRM + c] = b;
        tile[ty + 8 * i][tx] = b;
    }
    __syncthreads();
#pragma unroll
    for (int i = 0; i < 4; ++i) {
        int c = c0 + ty + 8 * i, r = r0 + tx;
        dstT[(size_t)c * R + r] = tile[tx][ty + 8 * i];
    }
}

// Plain f32 -> bf16 convert (weights: Bc[out][in] = W itself)
__global__ void convert_bf16(const float* __restrict__ src, bf16_t* __restrict__ dst, int n)
{
    int idx = blockIdx.x * blockDim.x + threadIdx.x;
    if (idx < n) dst[idx] = f2bf(src[idx]);
}

// rdeg_e: column sums, thread per column (coalesced across threads)
__global__ void compute_rdeg_e(const float* __restrict__ inc, float* __restrict__ rdeg_e)
{
    int e = blockIdx.x * blockDim.x + threadIdx.x;
    if (e >= EDG) return;
    float s = 0.0f;
    if (e < EMSK)
        for (int n = 0; n < NH; ++n) s += inc[(size_t)n * EDG + e];
    rdeg_e[e] = 1.0f / fmaxf(s, 1.0f);
}

// rdeg_v: one block per row, strided coalesced reduction
__global__ void __launch_bounds__(256)
compute_rdeg_v(const float* __restrict__ inc, float* __restrict__ rdeg_v)
{
    __shared__ float red[256];
    const int n = blockIdx.x;
    const int tid = threadIdx.x;
    float s = 0.0f;
    if (n < NH)
        for (int e = tid; e < EMSK; e += 256) s += inc[(size_t)n * EDG + e];
    red[tid] = s;
    __syncthreads();
    for (int o = 128; o > 0; o >>= 1) {
        if (tid < o) red[tid] += red[tid + o];
        __syncthreads();
    }
    if (tid == 0) rdeg_v[n] = 1.0f / fmaxf(red[0], 1.0f);
}

// ---------------------------------------------------------------------------
// Growth decision. node_mask = arange < N/2 ==> the k-th wanting parent (in
// index order) gets child slot NH + k; at most NH wants, exactly NH slots.
// ---------------------------------------------------------------------------
__global__ void grow_logits(const float* __restrict__ su,
                            const float* __restrict__ Wg,
                            const float* __restrict__ bg,
                            int* __restrict__ wants)
{
    int n = blockIdx.x * blockDim.x + threadIdx.x;
    if (n >= NTOT) return;
    float acc = bg[0];
    for (int d = 0; d < DIM; ++d) acc += su[(size_t)n * DIM + d] * Wg[d];
    float p = 1.0f / (1.0f + __expf(-acc));
    wants[n] = ((p > 0.5f) && (n < NH)) ? 1 : 0;
}

__global__ void __launch_bounds__(1024)
grow_scan(const int* __restrict__ wants, int* __restrict__ parent_of)
{
    __shared__ int part[1024];
    int tid = threadIdx.x;
    for (int i = tid; i < NTOT; i += 1024) parent_of[i] = -1;
    __syncthreads();

    int base = tid * 4;                      // 1024 * 4 = NH
    int w0 = wants[base + 0], w1 = wants[base + 1];
    int w2 = wants[base + 2], w3 = wants[base + 3];
    int cnt = w0 + w1 + w2 + w3;
    part[tid] = cnt;
    __syncthreads();
    for (int offd = 1; offd < 1024; offd <<= 1) {
        int v = (tid >= offd) ? part[tid - offd] : 0;
        __syncthreads();
        part[tid] += v;
        __syncthreads();
    }
    int r = part[tid] - cnt;                 // exclusive prefix
    if (w0) parent_of[NH + r++] = base + 0;
    if (w1) parent_of[NH + r++] = base + 1;
    if (w2) parent_of[NH + r++] = base + 2;
    if (w3) parent_of[NH + r]   = base + 3;
}

// ---------------------------------------------------------------------------
// Output assembly
// ---------------------------------------------------------------------------
__global__ void write_features(const float* __restrict__ nf,
                               const float* __restrict__ su,
                               const float* __restrict__ noise,
                               const int* __restrict__ parent_of,
                               float* __restrict__ out)
{
    int idx = blockIdx.x * blockDim.x + threadIdx.x;
    if (idx >= NTOT * DIM) return;
    int n = idx / DIM, d = idx % DIM;
    int p = parent_of[n];
    float v;
    if (p >= 0)   // newborn: parent's updated features + per-slot noise
        v = nf[(size_t)p * DIM + d] + su[(size_t)p * DIM + d] + noise[idx];
    else
        v = nf[idx] + ((n < NH) ? su[idx] : 0.0f);
    out[idx] = v;
}

__global__ void write_incidence(const float4* __restrict__ inc4,
                                const int* __restrict__ parent_of,
                                float4* __restrict__ out4)
{
    size_t idx = (size_t)blockIdx.x * blockDim.x + threadIdx.x;
    const int E4 = EDG / 4;
    if (idx >= (size_t)NTOT * E4) return;
    int n = (int)(idx / E4);
    int e4 = (int)(idx % E4);
    int p = parent_of[n];
    int row = (p >= 0) ? p : n;
    out4[idx] = inc4[(size_t)row * E4 + e4];
}

__global__ void write_masks(const int* __restrict__ parent_of,
                            float* __restrict__ out_nm,
                            float* __restrict__ out_em)
{
    int i = blockIdx.x * blockDim.x + threadIdx.x;
    if (i < NTOT) out_nm[i] = ((i < NH) || (parent_of[i] >= 0)) ? 1.0f : 0.0f;
    if (i < EDG)  out_em[i] = (i < EMSK) ? 1.0f : 0.0f;
}

// ---------------------------------------------------------------------------
extern "C" void kernel_launch(void* const* d_in, const int* in_sizes, int n_in,
                              void* d_out, int out_size, void* d_ws, size_t ws_size,
                              hipStream_t stream)
{
    const float* nf     = (const float*)d_in[0];
    const float* inc    = (const float*)d_in[1];
    const float* W_conv = (const float*)d_in[2];
    const float* b_conv = (const float*)d_in[3];
    const float* W1     = (const float*)d_in[4];
    const float* b1     = (const float*)d_in[5];
    const float* W2     = (const float*)d_in[6];
    const float* b2     = (const float*)d_in[7];
    const float* W3     = (const float*)d_in[8];
    const float* b3     = (const float*)d_in[9];
    const float* Wg     = (const float*)d_in[10];
    const float* bg     = (const float*)d_in[11];
    const float* noise  = (const float*)d_in[12];
    float* out = (float*)d_out;

    // Workspace carve-out (256B aligned)
    char* ws = (char*)d_ws;
    size_t off = 0;
    auto carve = [&](size_t bytes) {
        void* p = ws + off;
        off = (off + bytes + 255) & ~(size_t)255;
        return p;
    };
    bf16_t* Hm      = (bf16_t*)carve((size_t)NTOT * EDG * 2);   // masked incidence [N][E]
    bf16_t* HmT     = (bf16_t*)carve((size_t)NTOT * EDG * 2);   // transpose       [E][N]
    bf16_t* nfT     = (bf16_t*)carve((size_t)DIM * NTOT * 2);   // features^T      [D][N]
    bf16_t* emsgT   = (bf16_t*)carve((size_t)DIM * EDG * 2);    // edge messages^T [D][E]
    bf16_t* aggb    = (bf16_t*)carve((size_t)NTOT * DIM * 2);   // (Hm@emsg)/deg_v [N][D]
    bf16_t* xb      = (bf16_t*)carve((size_t)NTOT * 2 * DIM * 2); // concat input  [N][2D]
    bf16_t* h1      = (bf16_t*)carve((size_t)NTOT * HID * 2);
    bf16_t* h2      = (bf16_t*)carve((size_t)NTOT * HID * 2);
    float*  su      = (float*) carve((size_t)NTOT * DIM * 4);   // state_updates f32
    bf16_t* Wconv_b = (bf16_t*)carve((size_t)DIM * DIM * 2);    // col-major B == W
    bf16_t* W1_b    = (bf16_t*)carve((size_t)HID * 2 * DIM * 2);
    bf16_t* W2_b    = (bf16_t*)carve((size_t)HID * HID * 2);
    bf16_t* W3_b    = (bf16_t*)carve((size_t)DIM * HID * 2);
    float*  rdeg_e  = (float*) carve((size_t)EDG * 4);
    float*  rdeg_v  = (float*) carve((size_t)NTOT * 4);
    int*    wants   = (int*)   carve((size_t)NTOT * 4);
    int*    parent  = (int*)   carve((size_t)NTOT * 4);
    (void)ws_size; (void)in_sizes; (void)n_in; (void)out_size;

    // --- stage 0: conversions, transposes, degrees ---
    // masked incidence -> Hm (row-major) + HmT (transposed), both bf16
    tconv<<<dim3(EDG / 32, NTOT / 32), 256, 0, stream>>>(
        inc, NTOT, EDG, HmT, Hm, EDG, NH, EMSK);
    // node features -> nfT (transposed, for GEMM1 B) + xb[:, :D]
    tconv<<<dim3(DIM / 32, NTOT / 32), 256, 0, stream>>>(
        nf, NTOT, DIM, nfT, xb, 2 * DIM, NTOT, DIM);
    convert_bf16<<<(DIM * DIM) / 256, 256, 0, stream>>>(W_conv, Wconv_b, DIM * DIM);
    convert_bf16<<<(HID * 2 * DIM) / 256, 256, 0, stream>>>(W1, W1_b, HID * 2 * DIM);
    convert_bf16<<<(HID * HID) / 256, 256, 0, stream>>>(W2, W2_b, HID * HID);
    convert_bf16<<<(DIM * HID) / 256, 256, 0, stream>>>(W3, W3_b, DIM * HID);
    compute_rdeg_e<<<EDG / 256, 256, 0, stream>>>(inc, rdeg_e);
    compute_rdeg_v<<<NTOT, 256, 0, stream>>>(inc, rdeg_v);

    // --- stage 1: hypergraph conv (WMMA) ---
    // edge_msg^T = ((Hm^T @ X) / deg_e)^T   stored transposed -> col-major B for GEMM2
    wmma_gemm_bf16<<<dim3(EDG / 64, DIM / (16 * NT)), 128, 0, stream>>>(
        HmT, nfT, EDG, NTOT, DIM, rdeg_e, nullptr, 0,
        nullptr, 0, emsgT, EDG, 0, /*storeT=*/1);
    // agg = (Hm @ edge_msg) / deg_v
    wmma_gemm_bf16<<<dim3(NTOT / 64, DIM / (16 * NT)), 128, 0, stream>>>(
        Hm, emsgT, NTOT, EDG, DIM, rdeg_v, nullptr, 0,
        nullptr, 0, aggb, DIM, 0, 0);
    // x[:, D:2D] = agg @ W_conv.T + b_conv
    wmma_gemm_bf16<<<dim3(NTOT / 64, DIM / (16 * NT)), 128, 0, stream>>>(
        aggb, Wconv_b, NTOT, DIM, DIM, nullptr, b_conv, 0,
        nullptr, 0, xb, 2 * DIM, DIM, 0);

    // --- stage 2: MLP (WMMA) ---
    wmma_gemm_bf16<<<dim3(NTOT / 64, HID / (16 * NT)), 128, 0, stream>>>(
        xb, W1_b, NTOT, 2 * DIM, HID, nullptr, b1, 1,
        nullptr, 0, h1, HID, 0, 0);
    wmma_gemm_bf16<<<dim3(NTOT / 64, HID / (16 * NT)), 128, 0, stream>>>(
        h1, W2_b, NTOT, HID, HID, nullptr, b2, 1,
        nullptr, 0, h2, HID, 0, 0);
    wmma_gemm_bf16<<<dim3(NTOT / 64, DIM / (16 * NT)), 128, 0, stream>>>(
        h2, W3_b, NTOT, HID, DIM, nullptr, b3, 0,
        su, DIM, nullptr, 0, 0, 0);

    // --- stage 3: growth decision + output assembly ---
    grow_logits<<<NTOT / 256, 256, 0, stream>>>(su, Wg, bg, wants);
    grow_scan<<<1, 1024, 0, stream>>>(wants, parent);

    float* out_feat = out;
    float* out_inc  = out + (size_t)NTOT * DIM;
    float* out_nm   = out_inc + (size_t)NTOT * EDG;
    float* out_em   = out_nm + NTOT;
    write_features<<<(NTOT * DIM) / 256, 256, 0, stream>>>(nf, su, noise, parent, out_feat);
    write_incidence<<<(int)(((size_t)NTOT * (EDG / 4)) / 256), 256, 0, stream>>>(
        (const float4*)inc, parent, (float4*)out_inc);
    write_masks<<<NTOT / 256, 256, 0, stream>>>(parent, out_nm, out_em);
}
